// MultiHeadSelfAttentionWithRelativeBias_37125697307405
// MI455X (gfx1250) — compile-verified, matrix operand order checked by reading, not run on hardware
//
#include <hip/hip_runtime.h>

// ---------------------------------------------------------------------------
// MI455X (gfx1250) multi-head self-attention with relative bias.
// bf16 WMMA everywhere (v_wmma_f32_16x16x32_bf16), flash-style softmax,
// async global->LDS staging (GEMM tiles and attention K/V), DPP reductions.
// ---------------------------------------------------------------------------

typedef __bf16 bf16_t;
typedef bf16_t v16bf __attribute__((ext_vector_type(16)));
typedef bf16_t v8bf  __attribute__((ext_vector_type(8)));
typedef bf16_t v4bf  __attribute__((ext_vector_type(4)));
typedef float  v8f   __attribute__((ext_vector_type(8)));

// Types for the async-load builtin: param 1 is 'int __vector(4) __device__ *'
// (AS1 = global), param 2 is the LDS (AS3) counterpart.
typedef int v4i_vs __attribute__((vector_size(16)));
typedef __attribute__((address_space(1))) v4i_vs as1_v4i;
typedef __attribute__((address_space(3))) v4i_vs as3_v4i;

#define EMBED 1024
#define HEADS 16
#define HDIM  64
#define SEQ   1024
#define BATCH 16
#define MTOT  (BATCH * SEQ)  // 16384
#define CH    64             // attention key-chunk
#define LPAD  72             // padded LDS row (bf16): 144B, conflict-free
#define GPAD  40             // GEMM LDS row pad (bf16): 80B, conflict-free

#if __has_builtin(__builtin_amdgcn_global_load_async_to_lds_b128) && \
    __has_builtin(__builtin_amdgcn_s_wait_asynccnt)
#define HAVE_ASYNC_LDS 1
#else
#define HAVE_ASYNC_LDS 0
#endif

// Fragment loader: "one row per lane, k contiguous" WMMA operand pattern.
// Caller passes (row base + k0 + hf*8); per ISA 7.12.2 the lane needs
// k {0..7} at +0 and k {16..23} at +16 (halves swap for lanes 16..31).
static __device__ inline v16bf load_frag(const bf16_t* __restrict__ p) {
  v8bf lo = *(const v8bf*)p;
  v8bf hi = *(const v8bf*)(p + 16);
  return __builtin_shufflevector(lo, hi, 0, 1, 2, 3, 4, 5, 6, 7,
                                 8, 9, 10, 11, 12, 13, 14, 15);
}

static __device__ inline v8f wmma_bf16(v16bf a, v16bf b, v8f c) {
  return __builtin_amdgcn_wmma_f32_16x16x32_bf16(false, a, false, b,
                                                 (short)0, c, false, false);
}

// ---- DPP 16-lane butterfly allreduce (stays inside each row of 16) --------
template <int CTRL>
static __device__ inline float dpp_mov(float x) {
  int t = __builtin_amdgcn_update_dpp(__float_as_int(x), __float_as_int(x),
                                      CTRL, 0xf, 0xf, true);
  return __int_as_float(t);
}
static __device__ inline float row16_allmax(float x) {
  x = fmaxf(x, dpp_mov<0xB1>(x));   // quad_perm [1,0,3,2]  (xor 1)
  x = fmaxf(x, dpp_mov<0x4E>(x));   // quad_perm [2,3,0,1]  (xor 2)
  x = fmaxf(x, dpp_mov<0x141>(x));  // ROW_HALF_MIRROR      (xor 4 eff.)
  x = fmaxf(x, dpp_mov<0x140>(x));  // ROW_MIRROR           (xor 8 eff.)
  return x;
}
static __device__ inline float row16_allsum(float x) {
  x += dpp_mov<0xB1>(x);
  x += dpp_mov<0x4E>(x);
  x += dpp_mov<0x141>(x);
  x += dpp_mov<0x140>(x);
  return x;
}

// ---- async (or fallback) 16-byte global -> LDS stage ----------------------
static __device__ inline void stage_unit(const bf16_t* __restrict__ g,
                                         bf16_t* l) {
#if HAVE_ASYNC_LDS
  __builtin_amdgcn_global_load_async_to_lds_b128(
      (as1_v4i*)(uintptr_t)g, (as3_v4i*)(unsigned)(uintptr_t)l, 0, 0);
#else
  *(v8bf*)l = *(const v8bf*)g;
#endif
}
template <int N>
static __device__ inline void async_wait() {
#if HAVE_ASYNC_LDS
  __builtin_amdgcn_s_wait_asynccnt(N);
#endif
}

// ---------------------------------------------------------------------------
// fp32 -> bf16 convert, 4-wide
// ---------------------------------------------------------------------------
__global__ void cvt_bf16_kernel(const float* __restrict__ in,
                                bf16_t* __restrict__ out, int n4) {
  int i = blockIdx.x * blockDim.x + threadIdx.x;
  if (i < n4) {
    float4 f = ((const float4*)in)[i];
    v4bf o;
    o[0] = (bf16_t)f.x;
    o[1] = (bf16_t)f.y;
    o[2] = (bf16_t)f.z;
    o[3] = (bf16_t)f.w;
    ((v4bf*)out)[i] = o;
  }
}

// ---------------------------------------------------------------------------
// W [K][N] f32  ->  Wt [N][K] bf16  (so B-fragments load contiguously)
// ---------------------------------------------------------------------------
__global__ __launch_bounds__(256) void transpose_w_kernel(
    const float* __restrict__ in, bf16_t* __restrict__ out) {
  __shared__ float tile[32][33];
  const int bx = blockIdx.x * 32;
  const int by = blockIdx.y * 32;
  const int tx = threadIdx.x & 31;
  const int ty = threadIdx.x >> 5;
#pragma unroll
  for (int i = 0; i < 32; i += 8)
    tile[ty + i][tx] = in[(size_t)(by + ty + i) * EMBED + bx + tx];
  __syncthreads();
#pragma unroll
  for (int i = 0; i < 32; i += 8)
    out[(size_t)(bx + ty + i) * EMBED + by + tx] = (bf16_t)tile[tx][ty + i];
}

// ---------------------------------------------------------------------------
// GEMM: C[M=16384, N=1024] = A[M,K=1024] @ Wt^T   (A, Wt row-major bf16)
// Async double-buffered LDS staging of 128x32 A/B tiles (40 KB LDS).
// MODE 0: bf16 [B,H,S,D] (Q/K) | MODE 1: bf16 [B,H,D,S] (V^T) | MODE 2: f32
// Block: 128 threads = 4 waves 2x2, 64x64 per-wave tile (4x4 WMMA frags).
// ---------------------------------------------------------------------------
template <int MODE>
__global__ __launch_bounds__(128) void gemm_kernel(
    const bf16_t* __restrict__ A, const bf16_t* __restrict__ Bt,
    void* __restrict__ outp) {
  __shared__ __align__(16) bf16_t abuf[2][128][GPAD];  // 10 KB x2
  __shared__ __align__(16) bf16_t bbuf[2][128][GPAD];  // 10 KB x2

  const int lane = threadIdx.x & 31;
  const int wv = threadIdx.x >> 5;
  const int r = lane & 15;
  const int hf = lane >> 4;
  const int mw = (wv >> 1) * 64;  // wave tile inside block
  const int nw = (wv & 1) * 64;
  const int mb = blockIdx.y * 128;
  const int nb = blockIdx.x * 128;

  const v8f vz = {0.f, 0.f, 0.f, 0.f, 0.f, 0.f, 0.f, 0.f};
  v8f acc[4][4];
#pragma unroll
  for (int i = 0; i < 4; i++)
#pragma unroll
    for (int j = 0; j < 4; j++) acc[i][j] = vz;

  // Stage one 128x32 A tile + 128x32 B tile: 16 KB = 8 b128 units/thread.
  auto stage = [&](int kk, int bufi) {
#pragma unroll
    for (int s = 0; s < 4; s++) {
      const int u = threadIdx.x + s * 128;        // 0..511
      const int row = u >> 2, off = (u & 3) * 8;  // 4 units of 16B per row
      stage_unit(A + (size_t)(mb + row) * EMBED + kk + off,
                 &abuf[bufi][row][off]);
    }
#pragma unroll
    for (int s = 0; s < 4; s++) {
      const int u = threadIdx.x + s * 128;
      const int row = u >> 2, off = (u & 3) * 8;
      stage_unit(Bt + (size_t)(nb + row) * EMBED + kk + off,
                 &bbuf[bufi][row][off]);
    }
  };

  stage(0, 0);

  for (int ks = 0; ks < EMBED / 32; ks++) {
    const int bufi = ks & 1;
    if (ks + 1 < EMBED / 32) {
      stage((ks + 1) * 32, bufi ^ 1);  // prefetch next k-step
      async_wait<8>();                 // current k-step's 8 loads complete
    } else {
      async_wait<0>();
    }
    __syncthreads();

    v16bf af[4], bf_[4];
#pragma unroll
    for (int i = 0; i < 4; i++)
      af[i] = load_frag(&abuf[bufi][mw + i * 16 + r][hf * 8]);
#pragma unroll
    for (int j = 0; j < 4; j++)
      bf_[j] = load_frag(&bbuf[bufi][nw + j * 16 + r][hf * 8]);
#pragma unroll
    for (int i = 0; i < 4; i++)
#pragma unroll
      for (int j = 0; j < 4; j++)
        acc[i][j] = wmma_bf16(af[i], bf_[j], acc[i][j]);
    __syncthreads();  // tile consumed before its buffer is restaged
  }

#pragma unroll
  for (int i = 0; i < 4; i++) {
#pragma unroll
    for (int j = 0; j < 4; j++) {
#pragma unroll
      for (int v = 0; v < 8; v++) {
        const int m = mb + mw + i * 16 + v + 8 * hf;
        const int n = nb + nw + j * 16 + r;
        const float val = acc[i][j][v];
        if (MODE == 2) {
          ((float*)outp)[(size_t)m * EMBED + n] = val;
        } else {
          const int bb = m >> 10, s = m & 1023;
          const int h = n >> 6, d = n & 63;
          if (MODE == 0) {
            ((bf16_t*)outp)[((size_t)(bb * HEADS + h) * SEQ + s) * HDIM + d] =
                (bf16_t)val;
          } else {
            ((bf16_t*)outp)[((size_t)(bb * HEADS + h) * HDIM + d) * SEQ + s] =
                (bf16_t)val;
          }
        }
      }
    }
  }
}

// ---------------------------------------------------------------------------
// Flash attention per (b,h): 8 waves x 16 queries = 128 queries / block.
// K,V chunks (64 keys) staged to LDS with async b128 copies, double-buffered.
// Q,K: [B,H,S,D] bf16.  Vt: [B,H,D,S] bf16.  Out att: bf16 [B,S,E].
// ---------------------------------------------------------------------------
__global__ __launch_bounds__(256) void attn_kernel(
    const bf16_t* __restrict__ Q, const bf16_t* __restrict__ K,
    const bf16_t* __restrict__ Vt, const float* __restrict__ rel_bias,
    bf16_t* __restrict__ attout) {
  __shared__ float rbs[64];
  __shared__ __align__(16) bf16_t kbuf[2][CH][LPAD];  // 18 KB
  __shared__ __align__(16) bf16_t vbuf[2][CH][LPAD];  // 18 KB
  __shared__ __align__(16) bf16_t pbuf[8][16][LPAD];  // 18 KB

  const int bh = blockIdx.y;  // 0..255
  const int b = bh >> 4, h = bh & 15;
  const int lane = threadIdx.x & 31;
  const int wv = threadIdx.x >> 5;
  const int r = lane & 15;
  const int hf = lane >> 4;
  const int q0 = blockIdx.x * 128 + wv * 16;

  if (threadIdx.x < 63) rbs[threadIdx.x] = rel_bias[h * 63 + threadIdx.x];

  const bf16_t* __restrict__ Qb = Q + (size_t)bh * SEQ * HDIM;
  const bf16_t* __restrict__ Kb = K + (size_t)bh * SEQ * HDIM;
  const bf16_t* __restrict__ Vb = Vt + (size_t)bh * HDIM * SEQ;

  const v16bf qa0 = load_frag(Qb + (size_t)(q0 + r) * HDIM + hf * 8);
  const v16bf qa1 = load_frag(Qb + (size_t)(q0 + r) * HDIM + 32 + hf * 8);

  const v8f vz = {0.f, 0.f, 0.f, 0.f, 0.f, 0.f, 0.f, 0.f};
  v8f o[4] = {vz, vz, vz, vz};
  float mrun[8], lrun[8];
#pragma unroll
  for (int v = 0; v < 8; v++) {
    mrun[v] = -3.0e38f;
    lrun[v] = 0.f;
  }

  const float SCALE = 0.125f;  // 1/sqrt(64)
  const float L2E = 1.44269504088896340736f;

  // Cooperative stage of one 64-key chunk: K tile [64][64] + V tile [64][64],
  // 16 KB total = 4 b128 units per thread (256 threads).
  auto stage_chunk = [&](int c, int bufi) {
#pragma unroll
    for (int s = 0; s < 2; s++) {
      const int u = threadIdx.x + s * 256;        // 0..511
      const int row = u >> 3, off = (u & 7) * 8;  // 8 units of 16B per row
      stage_unit(Kb + (size_t)(c + row) * HDIM + off, &kbuf[bufi][row][off]);
    }
#pragma unroll
    for (int s = 0; s < 2; s++) {
      const int u = threadIdx.x + s * 256;
      const int row = u >> 3, off = (u & 7) * 8;
      stage_unit(Vb + (size_t)row * SEQ + c + off, &vbuf[bufi][row][off]);
    }
  };

  stage_chunk(0, 0);

  for (int ci = 0; ci < SEQ / CH; ci++) {
    const int c = ci * CH;
    const int bufi = ci & 1;
    if (ci + 1 < SEQ / CH) {
      stage_chunk(c + CH, bufi ^ 1);  // prefetch next chunk (other buffer)
      async_wait<4>();                // this chunk's 4 loads complete
    } else {
      async_wait<0>();
    }
    __syncthreads();  // all waves' staging for chunk ci visible

    // ---- scores: four 16-key tiles, contraction over D=64
    v8f s[4];
#pragma unroll
    for (int t = 0; t < 4; t++) {
      const v16bf ka = load_frag(&kbuf[bufi][t * 16 + r][hf * 8]);
      const v16bf kc = load_frag(&kbuf[bufi][t * 16 + r][32 + hf * 8]);
      s[t] = vz;
      s[t] = wmma_bf16(qa0, ka, s[t]);
      s[t] = wmma_bf16(qa1, kc, s[t]);
    }

    // ---- scale + decomposed relative bias (table in LDS)
#pragma unroll
    for (int t = 0; t < 4; t++) {
#pragma unroll
      for (int v = 0; v < 8; v++) {
        const int q = q0 + v + 8 * hf;
        const int ke = c + t * 16 + r;
        s[t][v] = s[t][v] * SCALE + rbs[(ke & 31) - (q & 31) + 31] +
                  rbs[(ke >> 5) - (q >> 5) + 31];
      }
    }

    // ---- online softmax over 64 keys (DPP reductions, no LDS traffic)
#pragma unroll
    for (int v = 0; v < 8; v++) {
      float tm = fmaxf(fmaxf(s[0][v], s[1][v]), fmaxf(s[2][v], s[3][v]));
      tm = row16_allmax(tm);
      const float mnew = fmaxf(mrun[v], tm);
      const float corr = __builtin_amdgcn_exp2f((mrun[v] - mnew) * L2E);
      float p[4], rs = 0.f;
#pragma unroll
      for (int t = 0; t < 4; t++) {
        p[t] = __builtin_amdgcn_exp2f((s[t][v] - mnew) * L2E);
        rs += p[t];
      }
      rs = row16_allsum(rs);
      lrun[v] = lrun[v] * corr + rs;
      mrun[v] = mnew;
#pragma unroll
      for (int f = 0; f < 4; f++) o[f][v] *= corr;
      const int m = v + 8 * hf;
#pragma unroll
      for (int t = 0; t < 4; t++) pbuf[wv][m][t * 16 + r] = (bf16_t)p[t];
    }

    // ---- reload P in A-frag layout; O += P @ V  (2 k-steps x 4 d-tiles)
    const v16bf pa0 = load_frag(&pbuf[wv][r][hf * 8]);
    const v16bf pa1 = load_frag(&pbuf[wv][r][32 + hf * 8]);
#pragma unroll
    for (int f = 0; f < 4; f++) {
      const v16bf vb0 = load_frag(&vbuf[bufi][f * 16 + r][hf * 8]);
      const v16bf vb1 = load_frag(&vbuf[bufi][f * 16 + r][32 + hf * 8]);
      o[f] = wmma_bf16(pa0, vb0, o[f]);
      o[f] = wmma_bf16(pa1, vb1, o[f]);
    }
    __syncthreads();  // chunk done before its buffer is restaged
  }

  // ---- normalize (v_rcp_f32) and store att [B,S,E] bf16
#pragma unroll
  for (int v = 0; v < 8; v++) {
    const float rinv = __builtin_amdgcn_rcpf(lrun[v]);
    const int q = q0 + v + 8 * hf;
#pragma unroll
    for (int f = 0; f < 4; f++) {
      const int d = f * 16 + r;
      attout[(size_t)(b * SEQ + q) * EMBED + h * HDIM + d] =
          (bf16_t)(o[f][v] * rinv);
    }
  }
}

// ---------------------------------------------------------------------------
// Host launcher. Workspace (72 MB):
//   [0,32MB): x_bf16 -> reused as att_bf16 | [32,40MB): 4 transposed weights
//   [40,72MB): V^T bf16.   Q -> d_out[0,32MB), K -> d_out[32,64MB).
// ---------------------------------------------------------------------------
extern "C" void kernel_launch(void* const* d_in, const int* in_sizes, int n_in,
                              void* d_out, int out_size, void* d_ws,
                              size_t ws_size, hipStream_t stream) {
  const float* x = (const float*)d_in[0];
  const float* Wq = (const float*)d_in[1];
  const float* Wk = (const float*)d_in[2];
  const float* Wv = (const float*)d_in[3];
  const float* Wo = (const float*)d_in[4];
  const float* rel_bias = (const float*)d_in[5];

  char* ws = (char*)d_ws;
  bf16_t* xb = (bf16_t*)(ws);
  bf16_t* att = xb;  // alias; x_bf16 dead before attention writes
  bf16_t* wqt = (bf16_t*)(ws + (size_t)32 * 1024 * 1024);
  bf16_t* wkt = wqt + (size_t)EMBED * EMBED;
  bf16_t* wvt = wkt + (size_t)EMBED * EMBED;
  bf16_t* wot = wvt + (size_t)EMBED * EMBED;
  bf16_t* Vt = (bf16_t*)(ws + (size_t)40 * 1024 * 1024);
  bf16_t* Qb = (bf16_t*)d_out;
  bf16_t* Kb = Qb + (size_t)MTOT * EMBED;

  const int n4 = MTOT * EMBED / 4;
  cvt_bf16_kernel<<<n4 / 256, 256, 0, stream>>>(x, xb, n4);

  dim3 tg(EMBED / 32, EMBED / 32);
  transpose_w_kernel<<<tg, 256, 0, stream>>>(Wq, wqt);
  transpose_w_kernel<<<tg, 256, 0, stream>>>(Wk, wkt);
  transpose_w_kernel<<<tg, 256, 0, stream>>>(Wv, wvt);
  transpose_w_kernel<<<tg, 256, 0, stream>>>(Wo, wot);

  dim3 gg(EMBED / 128, MTOT / 128);  // (8, 128)
  gemm_kernel<0><<<gg, 128, 0, stream>>>(xb, wqt, (void*)Qb);
  gemm_kernel<0><<<gg, 128, 0, stream>>>(xb, wkt, (void*)Kb);
  gemm_kernel<1><<<gg, 128, 0, stream>>>(xb, wvt, (void*)Vt);

  dim3 ag(SEQ / 128, BATCH * HEADS);  // (8, 256)
  attn_kernel<<<ag, 256, 0, stream>>>(Qb, Kb, Vt, rel_bias, att);

  gemm_kernel<2><<<gg, 128, 0, stream>>>(att, wot, d_out);
}
